// RLMALinear_4260607557799
// MI455X (gfx1250) — matrix operand order repaired and checked by reading, not compiled, not run on hardware
//
#include <hip/hip_runtime.h>
#include <stdint.h>

// ---------------------------------------------------------------------------
// RLMA Linear for gfx1250: out[m,o] = sum_d x[m,d] * W[o,d] + bias[o]
//   W = alpha * R(seed) + U @ V,  R = threefry-normal / sqrt(512)
// GEMM via V_WMMA_F32_16X16X32_BF16 with bf16 hi/lo split (fp32-class acc).
// B tile staged in LDS once per block via TDM (tensor_load_to_lds).
// ---------------------------------------------------------------------------

typedef __attribute__((ext_vector_type(16))) __bf16    v16bf;
typedef __attribute__((ext_vector_type(8)))  float     v8f;
typedef __attribute__((ext_vector_type(4)))  uint32_t  u32x4;
typedef __attribute__((ext_vector_type(8)))  uint32_t  u32x8;

union Frag16 { v16bf v; uint32_t u[8]; };
union Frag8F { v8f v; float f[8]; };

#define D_IN     512
#define D_OUT    512
#define RANK     64
#define NK_TILES 16            // 512 / 32 (K per WMMA)
#define NN_TILES 32            // 512 / 16
// Workspace: 8 N-groups; each group = contiguous 128 KB = [hi 64KB][lo 64KB].
// Within a plane, chunk (ntl,kt) = 1KB split as two 512B half-chunks
// (lane stride 16B) for conflict-friendly ds_load_b128.
#define NG_U32   32768         // u32 per N-group (128 KB)
#define PLANE_U32 16384        // u32 per plane (64 KB)

// ----------------------------- RNG (threefry2x32-20) -----------------------
__device__ __forceinline__ uint32_t rotl32(uint32_t v, int n) {
  return (v << n) | (v >> (32 - n));
}

__device__ __forceinline__ void threefry2x32(uint32_t k0, uint32_t k1,
                                             uint32_t x0, uint32_t x1,
                                             uint32_t& o0, uint32_t& o1) {
  uint32_t k2 = k0 ^ k1 ^ 0x1BD11BDAu;
  x0 += k0; x1 += k1;
#define TF_R(r) { x0 += x1; x1 = rotl32(x1, r); x1 ^= x0; }
  TF_R(13) TF_R(15) TF_R(26) TF_R(6)   x0 += k1; x1 += k2 + 1u;
  TF_R(17) TF_R(29) TF_R(16) TF_R(24)  x0 += k2; x1 += k0 + 2u;
  TF_R(13) TF_R(15) TF_R(26) TF_R(6)   x0 += k0; x1 += k1 + 3u;
  TF_R(17) TF_R(29) TF_R(16) TF_R(24)  x0 += k1; x1 += k2 + 4u;
  TF_R(13) TF_R(15) TF_R(26) TF_R(6)   x0 += k2; x1 += k0 + 5u;
#undef TF_R
  o0 = x0; o1 = x1;
}

// Giles (2012) single-precision inverse error function
__device__ __forceinline__ float erfinv_f(float x) {
  float w = -__logf((1.0f - x) * (1.0f + x));
  float p;
  if (w < 5.0f) {
    w -= 2.5f;
    p =              2.81022636e-08f;
    p = fmaf(p, w,   3.43273939e-07f);
    p = fmaf(p, w,  -3.52338770e-06f);
    p = fmaf(p, w,  -4.39150654e-06f);
    p = fmaf(p, w,   2.18580870e-04f);
    p = fmaf(p, w,  -1.25372503e-03f);
    p = fmaf(p, w,  -4.17768164e-03f);
    p = fmaf(p, w,   2.46640727e-01f);
    p = fmaf(p, w,   1.50140941e+00f);
  } else {
    w = sqrtf(w) - 3.0f;
    p =             -2.00214257e-04f;
    p = fmaf(p, w,   1.00950558e-04f);
    p = fmaf(p, w,   1.34934322e-03f);
    p = fmaf(p, w,  -3.67342844e-03f);
    p = fmaf(p, w,   5.73950773e-03f);
    p = fmaf(p, w,  -7.62246130e-03f);
    p = fmaf(p, w,   9.43887047e-03f);
    p = fmaf(p, w,   1.00167406e+00f);
    p = fmaf(p, w,   2.83297682e+00f);
  }
  return p * x;
}

__device__ __forceinline__ float synth_R(uint32_t seed, uint32_t idx) {
  const uint32_t half = (D_OUT * D_IN) / 2u;
  uint32_t c0, c1, o0, o1;
  if (idx < half) { c0 = idx;        c1 = idx + half; }
  else            { c0 = idx - half; c1 = idx;        }
  threefry2x32(0u, seed, c0, c1, o0, o1);
  uint32_t bits = (idx < half) ? o0 : o1;
  float f = __uint_as_float((bits >> 9) | 0x3f800000u) - 1.0f;  // [0,1)
  const float lo = -0.99999994f;                                 // nextafter(-1,0)
  float u = fmaf(f, 1.0f - lo, lo);
  u = fmaxf(u, lo);
  return 1.41421356237f * erfinv_f(u);
}

// pack {b.hi16, a.hi16} into one u32 with a single v_perm_b32
__device__ __forceinline__ uint32_t pack_hi16(uint32_t a, uint32_t b) {
  return __builtin_amdgcn_perm(b, a, 0x07060302u);
}

// ------------------- Kernel 1: build W in B-fragment order -----------------
// B[k,n] = W[n,k]. Fragment (ng,ntl,kt,lane): element j holds
// K = kt*32 + (lane/16)*16 + j, N = (ng*4+ntl)*16 + lane%16.
// Storage: group ng base = ng*NG_U32; plane hi at +0, lo at +PLANE_U32;
// chunk (ntl,kt) at +(ntl*16+kt)*256 u32; u[0..3] at lane*4, u[4..7] at
// 128 + lane*4 (two 512B half-chunks).
__global__ __launch_bounds__(256) void wgen_kernel(
    const float* __restrict__ alpha_p, const float* __restrict__ U,
    const float* __restrict__ V, const int* __restrict__ k_iter_p,
    uint32_t* __restrict__ wsB) {
  const int t    = blockIdx.x * 256 + threadIdx.x;   // 0..16383
  const int lane = t & 31;
  const int kt   = (t >> 5) & 15;
  const int nt   = t >> 9;                           // 0..31 global ntile
  const int ng   = nt >> 2;
  const int ntl  = nt & 3;
  const float    alpha = alpha_p[0];
  const uint32_t seed  = (1234u ^ (uint32_t)k_iter_p[0]);
  const int o     = nt * 16 + (lane & 15);
  const int dbase = kt * 32 + (lane >> 4) * 16;
  const float rscale = 0.0441941738241592f;          // 1/sqrt(512)

  float uv[16];
#pragma unroll
  for (int j = 0; j < 16; ++j) uv[j] = 0.0f;
  for (int r = 0; r < RANK; ++r) {
    const float u = U[o * RANK + r];
    const float4* vp = reinterpret_cast<const float4*>(V + r * D_IN + dbase);
    const float4 v0 = vp[0], v1 = vp[1], v2 = vp[2], v3 = vp[3];
    uv[0]  = fmaf(u, v0.x, uv[0]);  uv[1]  = fmaf(u, v0.y, uv[1]);
    uv[2]  = fmaf(u, v0.z, uv[2]);  uv[3]  = fmaf(u, v0.w, uv[3]);
    uv[4]  = fmaf(u, v1.x, uv[4]);  uv[5]  = fmaf(u, v1.y, uv[5]);
    uv[6]  = fmaf(u, v1.z, uv[6]);  uv[7]  = fmaf(u, v1.w, uv[7]);
    uv[8]  = fmaf(u, v2.x, uv[8]);  uv[9]  = fmaf(u, v2.y, uv[9]);
    uv[10] = fmaf(u, v2.z, uv[10]); uv[11] = fmaf(u, v2.w, uv[11]);
    uv[12] = fmaf(u, v3.x, uv[12]); uv[13] = fmaf(u, v3.y, uv[13]);
    uv[14] = fmaf(u, v3.z, uv[14]); uv[15] = fmaf(u, v3.w, uv[15]);
  }

  float w[16];
#pragma unroll
  for (int j = 0; j < 16; ++j) {
    const int d = dbase + j;
    const float Rv = synth_R(seed, (uint32_t)(o * D_IN + d)) * rscale;
    w[j] = fmaf(alpha, Rv, uv[j]);
  }

  const size_t chunk = (size_t)ng * NG_U32 + (size_t)(ntl * 16 + kt) * 256;
  uint32_t* hi = wsB + chunk;
  uint32_t* lo = wsB + chunk + PLANE_U32;
#pragma unroll
  for (int j = 0; j < 8; ++j) {
    const float a = w[2 * j], b = w[2 * j + 1];
    const uint32_t ab = __float_as_uint(a), bb = __float_as_uint(b);
    const float alr = a - __uint_as_float(ab & 0xFFFF0000u);
    const float blr = b - __uint_as_float(bb & 0xFFFF0000u);
    const uint32_t hv = pack_hi16(ab, bb);
    const uint32_t lv = pack_hi16(__float_as_uint(alr), __float_as_uint(blr));
    const int slot = (j < 4) ? (lane * 4 + j) : (128 + lane * 4 + (j - 4));
    hi[slot] = hv;
    lo[slot] = lv;
  }
}

// --------------------- Kernel 2: WMMA GEMM (bf16 split) --------------------
__global__ __launch_bounds__(256) void gemm_kernel(
    const float* __restrict__ x, const float* __restrict__ bias,
    const uint32_t* __restrict__ wsB, float* __restrict__ out) {
  extern __shared__ uint32_t smem[];   // 128 KB: [hi 64KB][lo 64KB]

  const int lane  = threadIdx.x & 31;
  const int wave  = threadIdx.x >> 5;
  const int khalf = lane >> 4;
  const int lrow  = lane & 15;
  const size_t m0 = (size_t)blockIdx.x * 128 + wave * 16;
  const int    n0 = blockIdx.y * 64;

  // ---- TDM: copy this block's 128 KB B tile (hi+lo) global -> LDS ----
  if (wave == 0) {
    const uint64_t ga =
        (uint64_t)(uintptr_t)(wsB + (size_t)blockIdx.y * NG_U32);
    const uint32_t lds_off = (uint32_t)(uintptr_t)smem;  // flat低32 = LDS offset
    // D# group 0: count=1 | lds_addr | global_addr[56:0] | type=2
    u32x4 g0 = { 1u,
                 lds_off,
                 (uint32_t)ga,
                 (uint32_t)((ga >> 32) & 0x01FFFFFFu) | 0x80000000u };
    // D# group 1: data_size=4B; tensor_dim0 = tile_dim0 = 32768; dim1 = 1
    u32x8 g1 = { 0x00020000u,          // [17:16] data_size = 2 (4 bytes)
                 0x80000000u,          // tensor_dim0[15:0]=0x8000 in [31:16]
                 0x00010000u,          // tensor_dim1[15:0]=1 in [31:16]
                 0x80000000u,          // tile_dim0=0x8000 in [31:16]
                 0x00000001u,          // tile_dim1=1, tile_dim2=0
                 (uint32_t)NG_U32,     // tensor_dim0_stride[31:0]
                 0u, 0u };
    u32x4 gz = { 0u, 0u, 0u, 0u };     // groups 2/3: dims 2..4 unused
    asm volatile("tensor_load_to_lds %0, %1, %2, %3"
                 :: "s"(g0), "s"(g1), "s"(gz), "s"(gz)
                 : "memory");
    __builtin_amdgcn_s_wait_tensorcnt(0);
  }
  __syncthreads();

  const float* xrow = x + (m0 + lrow) * D_IN;
  const uint32_t* sh = smem;               // hi plane
  const uint32_t* sl = smem + PLANE_U32;   // lo plane

  Frag8F acc[4];
#pragma unroll
  for (int i = 0; i < 4; ++i)
#pragma unroll
    for (int j = 0; j < 8; ++j) acc[i].f[j] = 0.0f;

  for (int kt = 0; kt < NK_TILES; ++kt) {
    // ---- A fragments: load 16 f32 of this row, split to bf16 hi/lo ----
    const int kb = kt * 32 + khalf * 16;
    const float4 c0 = *reinterpret_cast<const float4*>(xrow + kb);
    const float4 c1 = *reinterpret_cast<const float4*>(xrow + kb + 4);
    const float4 c2 = *reinterpret_cast<const float4*>(xrow + kb + 8);
    const float4 c3 = *reinterpret_cast<const float4*>(xrow + kb + 12);
    float xv[16] = {c0.x, c0.y, c0.z, c0.w, c1.x, c1.y, c1.z, c1.w,
                    c2.x, c2.y, c2.z, c2.w, c3.x, c3.y, c3.z, c3.w};
    Frag16 ah, al;
#pragma unroll
    for (int j = 0; j < 8; ++j) {
      const float a = xv[2 * j], b = xv[2 * j + 1];
      const uint32_t ab = __float_as_uint(a), bb = __float_as_uint(b);
      const float alr = a - __uint_as_float(ab & 0xFFFF0000u);
      const float blr = b - __uint_as_float(bb & 0xFFFF0000u);
      ah.u[j] = pack_hi16(ab, bb);
      al.u[j] = pack_hi16(__float_as_uint(alr), __float_as_uint(blr));
    }

    // ---- B fragments from LDS + 3 WMMAs per N tile ----
#pragma unroll
    for (int nt = 0; nt < 4; ++nt) {
      const int co = (nt * 16 + kt) * 256 + lane * 4;  // half-chunk slot
      Frag16 bh, bl;
      const uint4 h0 = *reinterpret_cast<const uint4*>(sh + co);
      const uint4 h1 = *reinterpret_cast<const uint4*>(sh + co + 128);
      const uint4 l0 = *reinterpret_cast<const uint4*>(sl + co);
      const uint4 l1 = *reinterpret_cast<const uint4*>(sl + co + 128);
      bh.u[0] = h0.x; bh.u[1] = h0.y; bh.u[2] = h0.z; bh.u[3] = h0.w;
      bh.u[4] = h1.x; bh.u[5] = h1.y; bh.u[6] = h1.z; bh.u[7] = h1.w;
      bl.u[0] = l0.x; bl.u[1] = l0.y; bl.u[2] = l0.z; bl.u[3] = l0.w;
      bl.u[4] = l1.x; bl.u[5] = l1.y; bl.u[6] = l1.z; bl.u[7] = l1.w;

      acc[nt].v = __builtin_amdgcn_wmma_f32_16x16x32_bf16(
          false, ah.v, false, bh.v, (short)0, acc[nt].v, false, false);
      acc[nt].v = __builtin_amdgcn_wmma_f32_16x16x32_bf16(
          false, ah.v, false, bl.v, (short)0, acc[nt].v, false, false);
      acc[nt].v = __builtin_amdgcn_wmma_f32_16x16x32_bf16(
          false, al.v, false, bh.v, (short)0, acc[nt].v, false, false);
    }
  }

  // ---- bias + store (C layout: M = v + 8*khalf, N = lrow) ----
#pragma unroll
  for (int nt = 0; nt < 4; ++nt) {
    const int col = n0 + nt * 16 + lrow;
    const float bv = bias[col];
#pragma unroll
    for (int v = 0; v < 8; ++v) {
      out[(m0 + khalf * 8 + v) * D_OUT + col] = acc[nt].f[v] + bv;
    }
  }
}

// ------------------------------- launcher ----------------------------------
extern "C" void kernel_launch(void* const* d_in, const int* in_sizes, int n_in,
                              void* d_out, int out_size, void* d_ws,
                              size_t ws_size, hipStream_t stream) {
  const float* x     = (const float*)d_in[0];
  const float* alpha = (const float*)d_in[1];
  const float* U     = (const float*)d_in[2];
  const float* V     = (const float*)d_in[3];
  const float* bias  = (const float*)d_in[4];
  const int*   kit   = (const int*)d_in[5];

  uint32_t* wsB = (uint32_t*)d_ws;   // 8 groups x 128 KB = 1 MB

  wgen_kernel<<<64, 256, 0, stream>>>(alpha, U, V, kit, wsB);

  const int M = 32 * 4096;                    // 131072 rows
  dim3 grid(M / 128, D_OUT / 64);             // 1024 x 8 blocks
  gemm_kernel<<<grid, 256, 131072, stream>>>(x, bias, wsB, (float*)d_out);
}
